// ShellConv_4990751998042
// MI455X (gfx1250) — matrix operand order repaired and checked by reading, not compiled, not run on hardware
//
#include <hip/hip_runtime.h>

// ---------------- problem dims ----------------
#define B_   8
#define N_   8192
#define M_   2048
#define K_   32
#define F1_  64
#define HID_ 32      // PF/2
#define PF_  64
#define CIN_ 128     // PF + F1
#define OUT_ 128
#define DIV_ 4
#define S_   8       // K/DIV, maxpool window
#define EPS_ 1e-5f

#define NPTS  (B_*M_*K_)        // 524288 neighbor points
#define NBM   (B_*M_)           // 16384 query rows
#define NPOOL (NBM*DIV_)        // 65536 pooled rows (count for BN3)
#define KDIM  (DIV_*CIN_)       // 512 = GEMM K

// ---------------- stats buffer layout (floats) ----------------
#define ST_S1 0      // 3  bn1 sum
#define ST_Q1 4      // 3  bn1 sumsq
#define ST_S2 8      // 32 bn2 sum
#define ST_Q2 40     // 32 bn2 sumsq
#define ST_S3 72     // 128 bn3 sum
#define ST_Q3 200    // 128 bn3 sumsq
#define ST_N  512

// ---------------- workspace layout (bytes) ----------------
#define OFF_STATS 0
#define OFF_IDX   2048
#define OFF_LOCAL (OFF_IDX   + NPTS*4)          //  2,099,200
#define OFF_POOL  (OFF_LOCAL + NPTS*3*4)        //  8,390,656
#define OFF_AH    (OFF_POOL  + NPOOL*CIN_*4)    // 41,945,088
#define OFF_BT    (OFF_AH    + NBM*KDIM*2)      // 58,722,304  (+128KB)

typedef __attribute__((ext_vector_type(16))) _Float16 v16h;
typedef __attribute__((ext_vector_type(8)))  _Float16 v8h;
typedef __attribute__((ext_vector_type(8)))  float    v8f;

// ================= 0: zero the stat accumulators =================
__global__ void zero_stats_kernel(float* __restrict__ stats) {
    stats[threadIdx.x] = 0.f;
}

// ================= 1: KNN (sorted 32-NN per query) =================
#define KNN_BQ   128
#define KNN_TILE 512
__global__ __launch_bounds__(KNN_BQ) void knn_kernel(
    const float* __restrict__ points, const float* __restrict__ queries,
    int* __restrict__ idx_out) {
  __shared__ float sdist[K_ * KNN_BQ];   // [slot][thread], stride avoids bank conflicts
  __shared__ int   sidx [K_ * KNN_BQ];
  __shared__ float ptile[KNN_TILE * 3];
  const int t = threadIdx.x;
  const int bpb = M_ / KNN_BQ;                        // 16 blocks per batch
  const int b = blockIdx.x / bpb;
  const int m = (blockIdx.x % bpb) * KNN_BQ + t;
  const float qx = queries[(b*M_ + m)*3 + 0];
  const float qy = queries[(b*M_ + m)*3 + 1];
  const float qz = queries[(b*M_ + m)*3 + 2];
  #pragma unroll
  for (int j = 0; j < K_; ++j) sdist[j*KNN_BQ + t] = 3.0e38f;
  float worst = 3.0e38f;
  for (int base = 0; base < N_; base += KNN_TILE) {
    __syncthreads();
    for (int i = t; i < KNN_TILE*3; i += KNN_BQ)
      ptile[i] = points[(b*N_ + base)*3 + i];
    __syncthreads();
    for (int p = 0; p < KNN_TILE; ++p) {
      float dx = qx - ptile[p*3+0];
      float dy = qy - ptile[p*3+1];
      float dz = qz - ptile[p*3+2];
      float d = dx*dx + dy*dy + dz*dz;
      if (d < worst) {                 // sorted insertion keeps nearest-first order
        int j = K_ - 1;
        while (j > 0 && sdist[(j-1)*KNN_BQ + t] > d) {
          sdist[j*KNN_BQ + t] = sdist[(j-1)*KNN_BQ + t];
          sidx [j*KNN_BQ + t] = sidx [(j-1)*KNN_BQ + t];
          --j;
        }
        sdist[j*KNN_BQ + t] = d;
        sidx [j*KNN_BQ + t] = base + p;
        worst = sdist[(K_-1)*KNN_BQ + t];
      }
    }
  }
  for (int j = 0; j < K_; ++j)
    idx_out[(b*M_ + m)*K_ + j] = sidx[j*KNN_BQ + t];
}

// ================= 2: local offsets + BN1 batch stats =================
__global__ __launch_bounds__(256) void local_stats_kernel(
    const float* __restrict__ points, const float* __restrict__ queries,
    const int* __restrict__ knn_idx, float* __restrict__ local_out,
    float* __restrict__ stats) {
  __shared__ float acc[6];
  const int t = threadIdx.x;
  if (t < 6) acc[t] = 0.f;
  __syncthreads();
  const int gid = blockIdx.x * 256 + t;         // < NPTS
  const int bm  = gid >> 5;
  const int b   = bm >> 11;                     // M_ = 2^11
  const int pi  = knn_idx[gid];
  const float l0 = queries[bm*3+0] - points[(b*N_ + pi)*3 + 0];
  const float l1 = queries[bm*3+1] - points[(b*N_ + pi)*3 + 1];
  const float l2 = queries[bm*3+2] - points[(b*N_ + pi)*3 + 2];
  local_out[gid*3+0] = l0; local_out[gid*3+1] = l1; local_out[gid*3+2] = l2;
  atomicAdd(&acc[0], l0);  atomicAdd(&acc[1], l1);  atomicAdd(&acc[2], l2);
  atomicAdd(&acc[3], l0*l0); atomicAdd(&acc[4], l1*l1); atomicAdd(&acc[5], l2*l2);
  __syncthreads();
  if (t < 3) atomicAdd(&stats[ST_S1 + t], acc[t]);
  else if (t < 6) atomicAdd(&stats[ST_Q1 + (t-3)], acc[t]);
}

// ================= 3: layer1 pass for BN2 batch stats (h1 not stored) =================
__global__ __launch_bounds__(256) void layer1_stats_kernel(
    const float* __restrict__ local_in, const float* __restrict__ w1,
    const float* __restrict__ b1, const float* __restrict__ g1,
    const float* __restrict__ be1, float* __restrict__ stats) {
  __shared__ float sw1[3*HID_], sb1[HID_];
  __shared__ float sc1[3], sh1[3];
  __shared__ float acc[2*HID_];
  const int t = threadIdx.x;
  if (t < 3*HID_) sw1[t] = w1[t];
  if (t < HID_) { sb1[t] = b1[t]; acc[t] = 0.f; acc[HID_+t] = 0.f; }
  if (t < 3) {
    const float inv = 1.f / (float)NPTS;
    const float mean = stats[ST_S1 + t] * inv;
    const float var  = stats[ST_Q1 + t] * inv - mean*mean;
    const float s = g1[t] * rsqrtf(var + EPS_);
    sc1[t] = s; sh1[t] = be1[t] - mean*s;
  }
  __syncthreads();
  const int gid = blockIdx.x * 256 + t;
  const float x0 = local_in[gid*3+0]*sc1[0] + sh1[0];
  const float x1 = local_in[gid*3+1]*sc1[1] + sh1[1];
  const float x2 = local_in[gid*3+2]*sc1[2] + sh1[2];
  #pragma unroll
  for (int j = 0; j < HID_; ++j) {
    float h = sb1[j] + x0*sw1[j] + x1*sw1[HID_+j] + x2*sw1[2*HID_+j];
    h = fmaxf(h, 0.f);
    atomicAdd(&acc[j], h);
    atomicAdd(&acc[HID_+j], h*h);
  }
  __syncthreads();
  if (t < 2*HID_) atomicAdd(&stats[ST_S2 + t], acc[t]);   // S2..Q2 contiguous
}

// ====== 4: fused layer1(recompute)+BN2+layer2+gather+concat+maxpool+BN3 stats ======
__global__ __launch_bounds__(256) void layer2_pool_kernel(
    const float* __restrict__ local_in, const int* __restrict__ knn_idx,
    const float* __restrict__ feat_prev,
    const float* __restrict__ w1, const float* __restrict__ b1,
    const float* __restrict__ w2, const float* __restrict__ b2,
    const float* __restrict__ g1, const float* __restrict__ be1,
    const float* __restrict__ g2, const float* __restrict__ be2,
    float* __restrict__ stats, float* __restrict__ pooled) {
  __shared__ float sw1[3*HID_], sb1[HID_];
  __shared__ float sw2[HID_*PF_], sb2[PF_];
  __shared__ float sc1[3], sh1v[3];
  __shared__ float sc2[HID_], sh2[HID_];
  __shared__ float h1n[K_*HID_];     // BN2-normalized layer1 output
  __shared__ float feat[K_*CIN_];    // [lift(64) | prev(64)]
  __shared__ float acc[2*CIN_];
  const int t  = threadIdx.x;
  const int bm = blockIdx.x;
  const int b  = bm >> 11;
  if (t < 3*HID_) sw1[t] = w1[t];
  if (t < HID_)   sb1[t] = b1[t];
  for (int i = t; i < HID_*PF_; i += 256) sw2[i] = w2[i];
  if (t < PF_)    sb2[t] = b2[t];
  if (t < 2*CIN_) acc[t] = 0.f;
  if (t < 3) {
    const float inv = 1.f / (float)NPTS;
    const float mean = stats[ST_S1 + t] * inv;
    const float var  = stats[ST_Q1 + t] * inv - mean*mean;
    const float s = g1[t] * rsqrtf(var + EPS_);
    sc1[t] = s; sh1v[t] = be1[t] - mean*s;
  }
  if (t < HID_) {
    const float inv = 1.f / (float)NPTS;
    const float mean = stats[ST_S2 + t] * inv;
    const float var  = stats[ST_Q2 + t] * inv - mean*mean;
    const float s = g2[t] * rsqrtf(var + EPS_);
    sc2[t] = s; sh2[t] = be2[t] - mean*s;
  }
  __syncthreads();
  // Phase A: h1n (32 pts x 32 feats), 4 entries/thread
  #pragma unroll
  for (int i = 0; i < 4; ++i) {
    const int e = t + i*256;
    const int k = e >> 5, j = e & (HID_-1);
    const int li = (bm*K_ + k)*3;
    const float x0 = local_in[li+0]*sc1[0] + sh1v[0];
    const float x1 = local_in[li+1]*sc1[1] + sh1v[1];
    const float x2 = local_in[li+2]*sc1[2] + sh1v[2];
    float h = sb1[j] + x0*sw1[j] + x1*sw1[HID_+j] + x2*sw1[2*HID_+j];
    h = fmaxf(h, 0.f);
    h1n[e] = h*sc2[j] + sh2[j];
  }
  __syncthreads();
  // Phase B: layer2 (32 pts x 64 feats), 8 entries/thread
  #pragma unroll
  for (int i = 0; i < 8; ++i) {
    const int e = t*8 + i;
    const int k = e >> 6, j = e & (PF_-1);
    float v = sb2[j];
    #pragma unroll
    for (int c = 0; c < HID_; ++c) v += h1n[k*HID_ + c] * sw2[c*PF_ + j];
    feat[k*CIN_ + j] = fmaxf(v, 0.f);
  }
  // Phase C: gather previous features into upper half (disjoint LDS region)
  #pragma unroll
  for (int i = 0; i < 8; ++i) {
    const int e = t*8 + i;
    const int k = e >> 6, c = e & (F1_-1);
    const int pi = knn_idx[bm*K_ + k];
    feat[k*CIN_ + PF_ + c] = feat_prev[(b*N_ + pi)*F1_ + c];
  }
  __syncthreads();
  // Phase D: maxpool window 8 along k; 2 outputs/thread; accumulate BN3 stats
  #pragma unroll
  for (int i = 0; i < 2; ++i) {
    const int o = t*2 + i;
    const int d = o >> 7, c = o & (CIN_-1);
    float mx = feat[(d*S_)*CIN_ + c];
    #pragma unroll
    for (int s = 1; s < S_; ++s) mx = fmaxf(mx, feat[(d*S_+s)*CIN_ + c]);
    pooled[(bm*DIV_ + d)*CIN_ + c] = mx;
    atomicAdd(&acc[c], mx);
    atomicAdd(&acc[CIN_ + c], mx*mx);
  }
  __syncthreads();
  if (t < 2*CIN_) atomicAdd(&stats[ST_S3 + t], acc[t]);   // S3..Q3 contiguous
}

// ================= 5: A = f16(BN3(pooled)), row = (b,m), col = d*128+c =================
__global__ __launch_bounds__(256) void abuild_kernel(
    const float* __restrict__ pooled, const float* __restrict__ stats,
    const float* __restrict__ g3, const float* __restrict__ be3,
    _Float16* __restrict__ ah) {
  const int i = blockIdx.x*256 + threadIdx.x;   // < NPOOL*CIN_
  const int c = i & (CIN_-1);
  const float inv = 1.f / (float)NPOOL;
  const float mean = stats[ST_S3 + c] * inv;
  const float var  = stats[ST_Q3 + c] * inv - mean*mean;
  const float sc = g3[c] * rsqrtf(var + EPS_);
  const float sh = be3[c] - mean*sc;
  ah[i] = (_Float16)(pooled[i]*sc + sh);
}

// ================= 6: Bt[o][d*128+c] = f16(wc[o][c][d]) =================
__global__ __launch_bounds__(256) void btbuild_kernel(
    const float* __restrict__ wc, _Float16* __restrict__ bt) {
  const int i = blockIdx.x*256 + threadIdx.x;   // < OUT_*KDIM
  const int o = i >> 9;
  const int r = i & (KDIM-1);
  const int d = r >> 7;
  const int c = r & (CIN_-1);
  bt[i] = (_Float16)wc[(o*CIN_ + c)*DIV_ + d];
}

// ===== 7: WMMA GEMM  out[16384x128] = relu(A[16384x512] @ Bt^T + bc)  =====
// One wave computes one 16x16 output tile; block of 8 waves covers all 8 col tiles.
__global__ __launch_bounds__(256) void conv_wmma_kernel(
    const _Float16* __restrict__ A, const _Float16* __restrict__ Bt,
    const float* __restrict__ bc, float* __restrict__ out) {
  const int lane = threadIdx.x & 31;
  const int wave = threadIdx.x >> 5;     // column tile 0..7
  const int row0 = blockIdx.x * 16;
  const int col0 = wave * 16;
  // A fragment map (16-bit A 16x32): lane<16 -> K 0..7,16..23; lane>=16 -> K 8..15,24..31
  const int arow = row0 + (lane & 15);
  const int akb  = (lane >> 4) * 8;
  // B fragment map (16-bit B 32x16): lane<16 -> K 0..7,16..23; lane>=16 -> K 8..15,24..31
  const int bcol = col0 + (lane & 15);
  const int bkb  = (lane >> 4) * 8;
  v8f accv = {};
  for (int kk = 0; kk < KDIM; kk += 32) {
    const _Float16* ap = A  + arow * KDIM + kk + akb;
    const _Float16* bp = Bt + bcol * KDIM + kk + bkb;
    const v8h alo = *(const v8h*)(ap);
    const v8h ahi = *(const v8h*)(ap + 16);
    const v8h blo = *(const v8h*)(bp);
    const v8h bhi = *(const v8h*)(bp + 16);
    v16h a, bfr;
    #pragma unroll
    for (int j = 0; j < 8; ++j) {
      a[j] = alo[j];  a[8+j] = ahi[j];
      bfr[j] = blo[j]; bfr[8+j] = bhi[j];
    }
    accv = __builtin_amdgcn_wmma_f32_16x16x32_f16(
        /*neg_a=*/false, a, /*neg_b=*/false, bfr,
        /*c_mod=*/(short)0, accv, /*reuse_a=*/false, /*reuse_b=*/false);
  }
  // C/D map: lane<16 -> N=lane, M=row0+r; lane>=16 -> N=lane-16, M=row0+8+r
  const int n = col0 + (lane & 15);
  const int rbase = row0 + (lane >> 4) * 8;
  const float bias = bc[n];
  #pragma unroll
  for (int r = 0; r < 8; ++r) {
    const float v = accv[r] + bias;
    out[(rbase + r) * OUT_ + n] = fmaxf(v, 0.f);
  }
}

// ================= host launcher =================
extern "C" void kernel_launch(void* const* d_in, const int* in_sizes, int n_in,
                              void* d_out, int out_size, void* d_ws, size_t ws_size,
                              hipStream_t stream) {
  (void)in_sizes; (void)n_in; (void)out_size; (void)ws_size;
  const float* points    = (const float*)d_in[0];
  const float* queries   = (const float*)d_in[1];
  const float* feat_prev = (const float*)d_in[2];
  const float* w1  = (const float*)d_in[3];
  const float* b1  = (const float*)d_in[4];
  const float* w2  = (const float*)d_in[5];
  const float* b2  = (const float*)d_in[6];
  const float* g1  = (const float*)d_in[7];
  const float* be1 = (const float*)d_in[8];
  const float* g2  = (const float*)d_in[9];
  const float* be2 = (const float*)d_in[10];
  const float* g3  = (const float*)d_in[11];
  const float* be3 = (const float*)d_in[12];
  const float* wc  = (const float*)d_in[13];
  const float* bc  = (const float*)d_in[14];
  float* out = (float*)d_out;

  char* ws = (char*)d_ws;
  float*    stats   = (float*)(ws + OFF_STATS);
  int*      knn_idx = (int*)  (ws + OFF_IDX);
  float*    local_b = (float*)(ws + OFF_LOCAL);
  float*    pooled  = (float*)(ws + OFF_POOL);
  _Float16* ah      = (_Float16*)(ws + OFF_AH);
  _Float16* bt      = (_Float16*)(ws + OFF_BT);

  zero_stats_kernel<<<1, ST_N, 0, stream>>>(stats);
  knn_kernel<<<B_*(M_/KNN_BQ), KNN_BQ, 0, stream>>>(points, queries, knn_idx);
  local_stats_kernel<<<NPTS/256, 256, 0, stream>>>(points, queries, knn_idx, local_b, stats);
  layer1_stats_kernel<<<NPTS/256, 256, 0, stream>>>(local_b, w1, b1, g1, be1, stats);
  layer2_pool_kernel<<<NBM, 256, 0, stream>>>(local_b, knn_idx, feat_prev,
      w1, b1, w2, b2, g1, be1, g2, be2, stats, pooled);
  btbuild_kernel<<<(OUT_*KDIM)/256, 256, 0, stream>>>(wc, bt);
  abuild_kernel<<<(NPOOL*CIN_)/256, 256, 0, stream>>>(pooled, stats, g3, be3, ah);
  conv_wmma_kernel<<<NBM/16, 256, 0, stream>>>(ah, bt, bc, out);
}
